// MultiScaleSTDF_22995254903519
// MI455X (gfx1250) — matrix operand order repaired and checked by reading, not compiled, not run on hardware
//
#include <hip/hip_runtime.h>
#include <hip/hip_bf16.h>

// ============================================================================
// MultiScaleSTDF forward for MI455X (gfx1250, wave32, WMMA).
//
// Unified implicit-GEMM conv (2D/3D) on v_wmma_f32_16x16x32_bf16.
// - zero-padded activation buffers  -> branch-free A fragments
// - LDS LUT of BYTE deltas          -> A address = one v_add (saddr+voffset)
// - readfirstlane'd tile coords     -> uniform SGPR base pointers
// - B fragment: 16 consecutive floats -> one voffset + immediate offsets
// - each wave: 32 pixels x 16 channels = 2 WMMAs/chunk sharing the B fragment
// - stride is a template param (constant-folded addressing)
// ============================================================================

typedef __attribute__((ext_vector_type(16))) __bf16 v16bf;
typedef __attribute__((ext_vector_type(8)))  float  v8f;

#define B_N   2
#define IN_C  7
#define OUT_C 64
#define NF    32
#define HH    160
#define WW    160

// ---------------------------------------------------------------------------
template <bool TAIL>
__device__ __forceinline__ void conv_chunk(
    v8f& a0, v8f& a1, const char* __restrict__ xc, const char* __restrict__ wc,
    int wb4, const int* lut, int kb, int hl, int base4a, int base4b, int KT)
{
  const int i0 = kb + hl * 16;
  int d[16];
  *(int4*)(d + 0)  = *(const int4*)(lut + i0);
  *(int4*)(d + 4)  = *(const int4*)(lut + i0 + 4);
  *(int4*)(d + 8)  = *(const int4*)(lut + i0 + 8);
  *(int4*)(d + 12) = *(const int4*)(lut + i0 + 12);

  float af0[16], af1[16], bf[16];
#pragma unroll
  for (int j = 0; j < 16; ++j) af0[j] = *(const float*)(xc + (base4a + d[j]));
#pragma unroll
  for (int j = 0; j < 16; ++j) af1[j] = *(const float*)(xc + (base4b + d[j]));

  const int kwb = kb + (hl << 4);
  const int vb  = wb4 + (kwb << 2);         // shared voffset; j*4 are immediates
  if (!TAIL) {
#pragma unroll
    for (int j = 0; j < 16; ++j) bf[j] = *(const float*)(wc + vb + (j << 2));
  } else {
#pragma unroll
    for (int j = 0; j < 16; ++j) {
      const int kw2 = kwb + j;
      const float vv = *(const float*)(wc + wb4 + ((kw2 < KT ? kw2 : 0) << 2));
      bf[j] = (kw2 < KT) ? vv : 0.f;
    }
  }

  v16bf av0, av1, bv;
#pragma unroll
  for (int j = 0; j < 16; ++j) {
    av0[j] = (__bf16)af0[j];
    av1[j] = (__bf16)af1[j];
    bv[j]  = (__bf16)bf[j];
  }
  a0 = __builtin_amdgcn_wmma_f32_16x16x32_bf16(false, av0, false, bv, (short)0, a0, false, false);
  a1 = __builtin_amdgcn_wmma_f32_16x16x32_bf16(false, av1, false, bv, (short)0, a1, false, false);
}

// ---------------------------------------------------------------------------
// Input PADDED (Dp,Hp,Wp), (oD,oH,oW) = border - pad.  Output logical coords
// shifted by (obD,obH,obW) inside padded (DpO,HpO,WpO).
// One wave: 32 consecutive output x  X  16 output channels.
// ---------------------------------------------------------------------------
template <int STRIDE>
__global__ __launch_bounds__(128) void convnd_wmma(
    const float* __restrict__ x, const float* __restrict__ w,
    const float* __restrict__ bias, const float* __restrict__ addsrc,
    float* __restrict__ y,
    int Bn, int Cin, int Dp, int Hp, int Wp, int oD, int oH, int oW,
    int Cout, int Dout, int Hout, int Wout,
    int DpO, int HpO, int WpO, int obD, int obH, int obW,
    int KD, int KH, int KW, int relu)
{
  __shared__ int lut[4096];
  const int KHW = KH * KW;
  const int KV  = KD * KHW;
  const int KT  = Cin * KV;                  // max 4000
  const int KTpad = (KT + 31) & ~31;
  for (int kk = threadIdx.x; kk < KTpad; kk += 128) {
    const int kkc = kk < KT ? kk : KT - 1;   // clamped tail: A harmless (B=0)
    const int c  = kkc / KV;  const int r  = kkc - c * KV;
    const int kd = r / KHW;   const int r2 = r - kd * KHW;
    const int ky = r2 / KW;   const int kx = r2 - ky * KW;
    lut[kk] = (((c * Dp + kd) * Hp + ky) * Wp + kx) << 2;   // BYTE delta
  }
  __syncthreads();

  const int lane = threadIdx.x & 31;
  const int hl   = lane >> 4;
  const int l16  = lane & 15;
  const int ntiles = (Cout + 15) >> 4;
  const int WT2    = (Wout + 31) >> 5;
  const long mtiles = (long)Bn * Dout * Hout * WT2;
  long wave = (long)blockIdx.x * 4 + (threadIdx.x >> 5);
  if (wave >= mtiles * (long)ntiles) return;       // wave-uniform exit

  int nt = (int)(wave % ntiles);
  long mt = wave / ntiles;
  int xt = (int)(mt % WT2); mt /= WT2;
  int oy = (int)(mt % Hout); mt /= Hout;
  int od = (int)(mt % Dout);
  int ob = (int)(mt / Dout);
  // force wave-uniform values into SGPRs -> uniform base pointers (saddr form)
  nt = __builtin_amdgcn_readfirstlane(nt);
  xt = __builtin_amdgcn_readfirstlane(xt);
  oy = __builtin_amdgcn_readfirstlane(oy);
  od = __builtin_amdgcn_readfirstlane(od);
  ob = __builtin_amdgcn_readfirstlane(ob);
  const int x0 = xt << 5;
  const int n0 = nt << 4;

  const int rowbase = ((od + oD) * Hp + oy * STRIDE + oH) * Wp + oW;
  const int oxA = min(x0 + l16,      Wout - 1);    // clamp overhang pixels
  const int oxB = min(x0 + 16 + l16, Wout - 1);
  const int base4a = (rowbase + oxA * STRIDE) << 2;
  const int base4b = (rowbase + oxB * STRIDE) << 2;
  const char* xc = (const char*)(x + (long)ob * Cin * Dp * Hp * Wp);
  const char* wc = (const char*)w;

  const int  n    = n0 + l16;
  const bool nok  = n < Cout;
  const int  wb4  = ((nok ? n : 0) * KT) << 2;

  v8f a0 = {}, a1 = {};
  const int KTmain = KT & ~31;
  for (int kb = 0; kb < KTmain; kb += 32)
    conv_chunk<false>(a0, a1, xc, wc, wb4, lut, kb, hl, base4a, base4b, KT);
  if (KTmain < KTpad)
    conv_chunk<true>(a0, a1, xc, wc, wb4, lut, KTmain, hl, base4a, base4b, KT);

  if (nok) {
    const float bb = bias[n];
    const long obase =
        ((((long)ob * Cout + n) * DpO + od + obD) * HpO + oy + obH) * (long)WpO + obW;
#pragma unroll
    for (int r = 0; r < 8; ++r) {
      const int oxp = x0 + hl * 8 + r;
      if (oxp < Wout) {
        float v = a0[r] + bb;
        if (addsrc) v += addsrc[obase + oxp];
        if (relu)   v = fmaxf(v, 0.f);
        y[obase + oxp] = v;
      }
      const int oxq = oxp + 16;
      if (oxq < Wout) {
        float v = a1[r] + bb;
        if (addsrc) v += addsrc[obase + oxq];
        if (relu)   v = fmaxf(v, 0.f);
        y[obase + oxq] = v;
      }
    }
  }
}

// ---------------------------------------------------------------------------
__global__ void pad_copy_kernel(const float* __restrict__ src, float* __restrict__ dst,
                                int Bn, int C, int Dn, int H, int W,
                                int Dp, int Hp, int Wp, int bD, int bH, int bW)
{
  const long total = (long)Bn * C * Dn * H * W;
  long idx = (long)blockIdx.x * blockDim.x + threadIdx.x;
  if (idx >= total) return;
  long t = idx;
  const int xx = (int)(t % W); t /= W;
  const int yy = (int)(t % H); t /= H;
  const int d  = (int)(t % Dn); t /= Dn;
  const int c  = (int)(t % C);
  const int n  = (int)(t / C);
  dst[((((long)n * C + c) * Dp + d + bD) * Hp + yy + bH) * (long)Wp + xx + bW] = src[idx];
}

// ---------------------------------------------------------------------------
// Transposed conv 4x4 stride2 (+ReLU).  In/out both border-1 padded.
// ---------------------------------------------------------------------------
__global__ void convT2d_kernel(const float* __restrict__ x, const float* __restrict__ w,
                               const float* __restrict__ bias, float* __restrict__ y,
                               int Bn, int C, int Hin, int Win)
{
  const int Hout = Hin * 2, Wout = Win * 2;
  const int Hp = Hin + 2,  Wp = Win + 2;
  const int HpO = Hout + 2, WpO = Wout + 2;
  const long total = (long)Bn * C * Hout * Wout;
  long idx = (long)blockIdx.x * blockDim.x + threadIdx.x;
  if (idx >= total) return;
  const int ox = (int)(idx % Wout); long t = idx / Wout;
  const int oy = (int)(t % Hout);   t /= Hout;
  const int o  = (int)(t % C);
  const int nb = (int)(t / C);
  const long chs = (long)Hp * Wp;
  const float* xb = x + (long)nb * C * chs;
  float s = bias[o];
  for (int a = 0; a < 4; ++a) {
    const int yy = oy - 2 + a;
    if (yy < 0 || (yy & 1)) continue;
    const int u = yy >> 1; if (u >= Hin) continue;
    for (int b2 = 0; b2 < 4; ++b2) {
      const int xx = ox - 2 + b2;
      if (xx < 0 || (xx & 1)) continue;
      const int v = xx >> 1; if (v >= Win) continue;
      const float* xp = xb + (long)(u + 1) * Wp + (v + 1);
      const float* wp = w + o * 16 + (3 - a) * 4 + (3 - b2);
      for (int i = 0; i < C; ++i)
        s += xp[(long)i * chs] * wp[(long)i * C * 16];
    }
  }
  y[((long)(nb * C + o) * HpO + oy + 1) * WpO + ox + 1] = fmaxf(s, 0.f);
}

// ---------------------------------------------------------------------------
__global__ void concat2_kernel(const float* __restrict__ a, const float* __restrict__ b,
                               float* __restrict__ o, int Bn, int C, long HW)
{
  const long total = (long)Bn * 2 * C * HW;
  long idx = (long)blockIdx.x * blockDim.x + threadIdx.x;
  if (idx >= total) return;
  const long per = 2 * (long)C * HW;
  const int n = (int)(idx / per);
  long r = idx - (long)n * per;
  const int c = (int)(r / HW);
  const long p = r - (long)c * HW;
  o[idx] = (c < C) ? a[((long)n * C + c) * HW + p]
                   : b[((long)n * C + (c - C)) * HW + p];
}

// mean over D=7: h (B,C,11,164,164) border-2 -> g (B,C,162,162) border-1
__global__ void mean_d_kernel(const float* __restrict__ h, float* __restrict__ g,
                              int Bn, int C)
{
  const long total = (long)Bn * C * HH * WW;
  long idx = (long)blockIdx.x * blockDim.x + threadIdx.x;
  if (idx >= total) return;
  long t = idx;
  const int xx = (int)(t % WW); t /= WW;
  const int yy = (int)(t % HH); t /= HH;
  const long nc = t;
  const long pl = (long)164 * 164;
  const float* hp = h + (nc * 11 + 2) * pl + (long)(yy + 2) * 164 + (xx + 2);
  float s = 0.f;
#pragma unroll
  for (int d = 0; d < 7; ++d) s += hp[d * pl];
  g[nc * (162L * 162) + (long)(yy + 1) * 162 + (xx + 1)] = s * (1.0f / 7.0f);
}

// ---------------------------------------------------------------------------
// Deformable conv (reads UNPADDED x and om).
// ---------------------------------------------------------------------------
__global__ __launch_bounds__(256) void deform_conv_kernel(
    const float* __restrict__ x, const float* __restrict__ om,
    const float* __restrict__ w, const float* __restrict__ bias,
    float* __restrict__ out, int Bn, int Hn, int Wn)
{
  __shared__ float sw[OUT_C * IN_C * 9];
  __shared__ float ss[32 * IN_C * 9];
  const int tid = threadIdx.x;
  for (int i = tid; i < OUT_C * IN_C * 9; i += 256) sw[i] = w[i];

  const long HW   = (long)Hn * Wn;
  const long npix = (long)Bn * HW;
  const long pix0 = (long)blockIdx.x * 32;

  for (int t = tid; t < 32 * 63; t += 256) {
    const int p = t / 63, j = t - p * 63;
    float val = 0.f;
    const long pix = pix0 + p;
    if (pix < npix) {
      const int n = (int)(pix / HW);
      const long rem = pix - (long)n * HW;
      const int oy = (int)(rem / Wn);
      const int ox = (int)(rem - (long)oy * Wn);
      const int c = j / 9, k = j - c * 9;
      const int ky = k / 3 - 1, kx = k % 3 - 1;
      const float* omb = om + (long)n * (IN_C * 27) * HW;
      const float offy = omb[(long)(c * 18 + k * 2    ) * HW + rem];
      const float offx = omb[(long)(c * 18 + k * 2 + 1) * HW + rem];
      float m          = omb[(long)(126 + c * 9 + k   ) * HW + rem];
      m = 1.f / (1.f + __expf(-m));
      const float py = (float)oy + (float)ky + offy;
      const float px = (float)ox + (float)kx + offx;
      const float fy = floorf(py), fx = floorf(px);
      const int  y0 = (int)fy,  x0 = (int)fx;
      const float wy1 = py - fy, wx1 = px - fx;
      const float wy0 = 1.f - wy1, wx0 = 1.f - wx1;
      const float* xcp = x + ((long)n * IN_C + c) * HW;
      float v = 0.f;
#pragma unroll
      for (int dy = 0; dy < 2; ++dy)
#pragma unroll
        for (int dx = 0; dx < 2; ++dx) {
          const int yy = y0 + dy, xx = x0 + dx;
          if (yy >= 0 && yy < Hn && xx >= 0 && xx < Wn)
            v += (dy ? wy1 : wy0) * (dx ? wx1 : wx0) * xcp[(long)yy * Wn + xx];
        }
      val = v * m;
    }
    ss[p * 63 + j] = val;
  }
  __syncthreads();

  for (int t = tid; t < 32 * OUT_C; t += 256) {
    const int p = t >> 6, o = t & 63;
    const long pix = pix0 + p;
    if (pix >= npix) continue;
    const int n = (int)(pix / HW);
    const long rem = pix - (long)n * HW;
    float s = bias[o];
    const float* sp = ss + p * 63;
    const float* wp = sw + o * 63;
#pragma unroll 9
    for (int j = 0; j < 63; ++j) s += sp[j] * wp[j];
    out[((long)n * OUT_C + o) * HW + rem] = fmaxf(s, 0.f);
  }
}

// ============================================================================
extern "C" void kernel_launch(void* const* d_in, const int* in_sizes, int n_in,
                              void* d_out, int out_size, void* d_ws, size_t ws_size,
                              hipStream_t stream)
{
  (void)in_sizes; (void)n_in; (void)out_size;

  const float* x      = (const float*)d_in[0];
  const float* in_w   = (const float*)d_in[1];
  const float* in_b   = (const float*)d_in[2];
  const float* dn1_w1 = (const float*)d_in[3];
  const float* dn1_b1 = (const float*)d_in[4];
  const float* dn1_w2 = (const float*)d_in[5];
  const float* dn1_b2 = (const float*)d_in[6];
  const float* up1_w  = (const float*)d_in[7];
  const float* up1_b  = (const float*)d_in[8];
  const float* up1_wt = (const float*)d_in[9];
  const float* up1_bt = (const float*)d_in[10];
  const float* dn2_w1 = (const float*)d_in[11];
  const float* dn2_b1 = (const float*)d_in[12];
  const float* dn2_w2 = (const float*)d_in[13];
  const float* dn2_b2 = (const float*)d_in[14];
  const float* up2_w  = (const float*)d_in[15];
  const float* up2_b  = (const float*)d_in[16];
  const float* up2_wt = (const float*)d_in[17];
  const float* up2_bt = (const float*)d_in[18];
  const float* tr_w1  = (const float*)d_in[19];
  const float* tr_b1  = (const float*)d_in[20];
  const float* tr_w2  = (const float*)d_in[21];
  const float* tr_b2  = (const float*)d_in[22];
  const float* tr_wt  = (const float*)d_in[23];
  const float* tr_bt  = (const float*)d_in[24];
  const float* out_w  = (const float*)d_in[25];
  const float* out_b  = (const float*)d_in[26];
  const float* b3_w1  = (const float*)d_in[27];
  const float* b3_b1  = (const float*)d_in[28];
  const float* b3_w2  = (const float*)d_in[29];
  const float* b3_b2  = (const float*)d_in[30];
  const float* b5_w1  = (const float*)d_in[31];
  const float* b5_b1  = (const float*)d_in[32];
  const float* b5_w2  = (const float*)d_in[33];
  const float* b5_b2  = (const float*)d_in[34];
  const float* tf_w   = (const float*)d_in[35];
  const float* tf_b   = (const float*)d_in[36];
  const float* om_w   = (const float*)d_in[37];
  const float* om_b   = (const float*)d_in[38];
  const float* dcn_w  = (const float*)d_in[39];
  const float* dcn_b  = (const float*)d_in[40];
  float* outp = (float*)d_out;
  float* ws   = (float*)d_ws;

  // -------- workspace layout (floats); padded buffers; peak ~184 MB --------
  float* F0   = ws + 0;          // f0 -> t7        (1,679,616 = 2*32*162*162)
  float* F1   = ws + 1679616;    // f1              (  430,336 = 2*32*82*82)
  float* F2   = ws + 2109952;    // f2              (  112,896 = 2*32*42*42)
  float* S1   = ws + 2222848;    // d1a->t5->t6     (  430,336)
  float* S2   = ws + 2653184;    // d2a->cat1       (  225,792 = 2*64*42*42)
  float* S3   = ws + 2878976;    // t3->t4          (  112,896)
  float* T20A = ws + 2991872;    //                 (   30,976 = 2*32*22*22)
  float* T20B = ws + 3022848;    //                 (   30,976)
  float* CAT2 = ws + 3053824;    //                 (  860,672 = 2*64*82*82)
  float* G3   = ws + 3914496;    //                 (1,679,616)
  float* G5   = ws + 5594112;    //                 (1,679,616)
  float* XP1  = ws + 7273728;    // x, border1      (  367,416 = 2*7*162*162)
  float* XP3  = ws + 7641144;    // x, 3D border2   (  591,712 = 2*11*164*164)
  float* HB1  = ws + 8232856;    // 3D buf A        (18,934,784 = 2*32*11*164*164)
  float* HB2  = ws + 27167640;   // 3D buf B        (18,934,784)
  float* CAT3 = HB1;                   // 3,359,232 = 2*64*162*162
  float* FUSE = HB1 + 3359232;         // 1,679,616
  float* O1   = HB1 + 5038848;         // 1,679,616
  float* OM   = HB1 + 6718464;         // 9,676,800 = 2*189*160*160 (unpadded)
  const size_t WS_FLOATS = 46102424;

  size_t zb = WS_FLOATS * sizeof(float);
  if (zb > ws_size) zb = ws_size;
  hipMemsetAsync(d_ws, 0, zb, stream);   // zero borders (graph-capturable)

  auto conv = [&](const float* xi, const float* wi, const float* bi, const float* add,
                  float* yo, int Cin, int Dp, int Hp, int Wp, int oD, int oH, int oW,
                  int Cout, int Dout, int Hout, int Wout,
                  int DpO, int HpO, int WpO, int obD, int obH, int obW,
                  int KD, int KH, int KW, int stride, int relu) {
    const long waves = (long)((Cout + 15) / 16) * B_N * Dout * Hout * ((Wout + 31) / 32);
    const int blocks = (int)((waves + 3) / 4);
    if (stride == 2)
      convnd_wmma<2><<<blocks, 128, 0, stream>>>(xi, wi, bi, add, yo,
          B_N, Cin, Dp, Hp, Wp, oD, oH, oW, Cout, Dout, Hout, Wout,
          DpO, HpO, WpO, obD, obH, obW, KD, KH, KW, relu);
    else
      convnd_wmma<1><<<blocks, 128, 0, stream>>>(xi, wi, bi, add, yo,
          B_N, Cin, Dp, Hp, Wp, oD, oH, oW, Cout, Dout, Hout, Wout,
          DpO, HpO, WpO, obD, obH, obW, KD, KH, KW, relu);
  };
  auto conv2 = [&](const float* xi, const float* wi, const float* bi, const float* add,
                   float* yo, int Cin, int Hin, int Cout, int stride, int relu, int obd) {
    const int Hout = (Hin - 1) / stride + 1;
    conv(xi, wi, bi, add, yo, Cin, 1, Hin + 2, Hin + 2, 0, 0, 0,
         Cout, 1, Hout, Hout, 1, Hout + 2 * obd, Hout + 2 * obd, 0, obd, obd,
         1, 3, 3, stride, relu);
  };
  auto conv3 = [&](const float* xi, const float* wi, const float* bi, float* yo,
                   int Cin, int K, int o) {
    conv(xi, wi, bi, nullptr, yo, Cin, 11, 164, 164, o, o, o,
         NF, IN_C, HH, WW, 11, 164, 164, 2, 2, 2, K, K, K, 1, 1);
  };
  auto convT = [&](const float* xi, const float* wi, const float* bi, float* yo, int Hin) {
    const long total = (long)B_N * NF * (2 * Hin) * (2 * Hin);
    convT2d_kernel<<<(int)((total + 255) / 256), 256, 0, stream>>>(
        xi, wi, bi, yo, B_N, NF, Hin, Hin);
  };
  auto concat = [&](const float* a, const float* b, float* o, long HWp) {
    const long total = (long)B_N * 2 * NF * HWp;
    concat2_kernel<<<(int)((total + 255) / 256), 256, 0, stream>>>(a, b, o, B_N, NF, HWp);
  };
  auto padcpy = [&](const float* s, float* dd, int C, int Dn, int Dp, int Hp,
                    int bD, int bHW) {
    const long total = (long)B_N * C * Dn * HH * WW;
    pad_copy_kernel<<<(int)((total + 255) / 256), 256, 0, stream>>>(
        s, dd, B_N, C, Dn, HH, WW, Dp, Hp, Hp, bD, bHW, bHW);
  };

  // ---- padded copies of the input ----
  padcpy(x, XP1, IN_C, 1, 1, 162, 0, 1);      // (B,7,162,162)
  padcpy(x, XP3, 1, IN_C, 11, 164, 2, 2);     // (B,1,11,164,164)

  // ---- U-Net trunk ----
  conv2(XP1,  in_w,  in_b,  nullptr, F0,  IN_C, 160, NF, 1, 1, 1);   // f0
  conv2(F0,  dn1_w1, dn1_b1, nullptr, S1,  NF,  160, NF, 2, 1, 1);
  conv2(S1,  dn1_w2, dn1_b2, nullptr, F1,  NF,   80, NF, 1, 1, 1);   // f1
  conv2(F1,  dn2_w1, dn2_b1, nullptr, S2,  NF,   80, NF, 2, 1, 1);
  conv2(S2,  dn2_w2, dn2_b2, nullptr, F2,  NF,   40, NF, 1, 1, 1);   // f2
  conv2(F2,  tr_w1,  tr_b1,  nullptr, T20A, NF,  40, NF, 2, 1, 1);
  conv2(T20A, tr_w2, tr_b2,  nullptr, T20B, NF,  20, NF, 1, 1, 1);
  convT(T20B, tr_wt, tr_bt,  S3, 20);                                // -> 40
  concat(S3, F2, S2, 42L * 42);                                      // cat1
  conv2(S2,  up2_w,  up2_b,  nullptr, S3, 2 * NF, 40, NF, 1, 1, 1);  // t4
  convT(S3,  up2_wt, up2_bt, S1, 40);                                // t5 -> 80
  concat(S1, F1, CAT2, 82L * 82);                                    // cat2
  conv2(CAT2, up1_w, up1_b,  nullptr, S1, 2 * NF, 80, NF, 1, 1, 1);  // t6
  convT(S1,  up1_wt, up1_bt, F0, 80);                                // t7 -> 160

  // ---- 3D temporal branches ----
  conv3(XP3, b3_w1, b3_b1, HB1, 1,  3, 1);
  conv3(HB1, b3_w2, b3_b2, HB2, NF, 3, 1);
  mean_d_kernel<<<(int)(((long)B_N * NF * HH * WW + 255) / 256), 256, 0, stream>>>(
      HB2, G3, B_N, NF);
  conv3(XP3, b5_w1, b5_b1, HB1, 1,  5, 0);
  conv3(HB1, b5_w2, b5_b2, HB2, NF, 5, 0);
  mean_d_kernel<<<(int)(((long)B_N * NF * HH * WW + 255) / 256), 256, 0, stream>>>(
      HB2, G5, B_N, NF);

  // ---- fuse, offset/mask head, deformable conv ----
  concat(G3, G5, CAT3, 162L * 162);
  conv2(CAT3, tf_w, tf_b, /*add=*/F0, FUSE, 2 * NF, 160, NF, 1, 1, 1); // relu(t+tf)
  conv2(FUSE, out_w, out_b, nullptr, O1, NF, 160, NF, 1, 1, 1);
  conv2(O1,   om_w,  om_b,  nullptr, OM, NF, 160, IN_C * 27, 1, 0, 0); // 189ch
  deform_conv_kernel<<<(int)(((long)B_N * HH * WW + 31) / 32), 256, 0, stream>>>(
      x, OM, dcn_w, dcn_b, outp, B_N, HH, WW);
}
